// GTN_grok_40132174414148
// MI455X (gfx1250) — matrix operand order repaired and checked by reading, not compiled, not run on hardware
//
#include <hip/hip_runtime.h>
#include <hip/hip_bf16.h>

typedef __attribute__((ext_vector_type(16))) _Float16 v16h;
typedef __attribute__((ext_vector_type(8)))  float    v8f;
typedef __attribute__((ext_vector_type(4)))  int      v4i;

typedef __attribute__((address_space(1))) v4i gv4i;  // global int4
typedef __attribute__((address_space(3))) v4i lv4i;  // LDS int4

#if defined(__has_builtin)
#if __has_builtin(__builtin_amdgcn_global_load_async_to_lds_b128)
#define GTN_HAVE_ASYNC_LDS 1
#endif
#endif
#ifndef GTN_HAVE_ASYNC_LDS
#define GTN_HAVE_ASYNC_LDS 0
#endif

namespace {
constexpr int kN  = 50000;   // nodes
constexpr int kE  = 800000;  // edges per layer
constexpr int kD  = 128;     // embed dim
constexpr int kH  = 8;       // heads
constexpr int kDH = 16;      // head dim
constexpr int kL  = 2;       // layers
constexpr int kP  = 8192;    // pos/neg pairs
constexpr int kFF = 256;     // FFN hidden
}

enum { GMODE_BIAS = 0, GMODE_RELU = 1, GMODE_LN = 2 };

__device__ __forceinline__ void gtn_async_cp16(const float* g, float* l) {
#if GTN_HAVE_ASYNC_LDS
  __builtin_amdgcn_global_load_async_to_lds_b128((gv4i*)g, (lv4i*)l, 0, 0);
#else
  *(float4*)l = *(const float4*)g;
#endif
}

__device__ __forceinline__ void gtn_wait_async() {
#if GTN_HAVE_ASYNC_LDS
  asm volatile("s_wait_asynccnt 0x0" ::: "memory");
#endif
}

// ---------------------------------------------------------------------------
// Fused GEMM: out[M x NC] = A[M x K] @ W[K x NC] + bias (+relu | +resid,LN)
// 256 threads = 8 wave32; block owns 128 rows.
//  - W staged once per block into LDS pre-swizzled into WMMA B-fragment order.
//  - A streamed 16 rows at a time into a double-buffered raw-f32 LDS tile via
//    global_load_async_to_lds_b128 (prefetching subtile s+1 while computing s),
//    then each lane reads its two contiguous 32B runs and converts f32->f16
//    in registers for the A fragment.
//  - Inner loop is ds_load_b128 -> v_wmma_f32_16x16x32_f16 (f32 accumulate).
// ---------------------------------------------------------------------------
template<int K, int NC, int MODE>
__global__ __launch_bounds__(256) void gtn_gemm(
    const float* __restrict__ A, const float* __restrict__ W,
    const float* __restrict__ bias, const float* __restrict__ resid,
    const float* __restrict__ lng, const float* __restrict__ lnb,
    float* __restrict__ out, int M)
{
  constexpr int KC  = K / 32;        // 32-wide K chunks
  constexpr int NT  = NC / 16;       // 16-wide column tiles
  constexpr int TPW = (NT + 7) / 8;  // column tiles per wave (1 or 2)
  constexpr int AP  = K + 4;         // padded A row stride (floats): 528B/1040B
                                     // -> rows spread over the 64 LDS banks

  __shared__ alignas(16) float    sRawA[2][16][AP];
  __shared__ alignas(32) _Float16 sB[KC][NT][32][16];
  __shared__ float sO[(MODE == GMODE_LN) ? 16 : 1][(MODE == GMODE_LN) ? NC : 1];
  __shared__ float sSum[(MODE == GMODE_LN) ? 16 : 1][16];
  __shared__ float sSq [(MODE == GMODE_LN) ? 16 : 1][16];
  __shared__ float sMean[(MODE == GMODE_LN) ? 16 : 1];
  __shared__ float sRstd[(MODE == GMODE_LN) ? 16 : 1];

  const int tid  = threadIdx.x;
  const int lane = tid & 31;
  const int wave = tid >> 5;
  const int blockRow0 = blockIdx.x * 128;

  // Async-stage one 16xK f32 A subtile into sRawA[buf] (16B granules).
  auto issueA = [&](int s, int buf) {
    const int r0 = blockRow0 + s * 16;
    for (int i = tid; i < 16 * (K / 4); i += 256) {
      const int m = i / (K / 4), cv = i % (K / 4);
      gtn_async_cp16(&A[(size_t)(r0 + m) * K + cv * 4], &sRawA[buf][m][cv * 4]);
    }
  };

  // Stage W -> LDS in B-fragment order: B element (k,n) lives at
  //   chunk c=k/32, tile t=n/16, lane=(n%16)+16*((k%32)/16), half j=k%16.
  for (int i = tid; i < K * NC; i += 256) {
    const int k = i / NC, n = i % NC;
    const int c = k >> 5, kk = k & 31;
    const int t  = n >> 4;
    const int ln = (n & 15) + ((kk >> 4) << 4);
    const int j  = kk & 15;
    sB[c][t][ln][j] = (_Float16)W[i];
  }

  int buf = 0;
  issueA(0, 0);

  for (int s = 0; s < 8; ++s) {
    const int row0 = blockRow0 + s * 16;
    if (row0 >= M) break;
    gtn_wait_async();   // own async loads for subtile s complete
    __syncthreads();    // everyone's async data visible; prev subtile reads done
    if (s < 7 && row0 + 16 < M) issueA(s + 1, buf ^ 1);  // overlap DMA w/ WMMA

    v8f acc[TPW];
    #pragma unroll
    for (int u = 0; u < TPW; ++u)
      #pragma unroll
      for (int r = 0; r < 8; ++r) acc[u][r] = 0.f;

    const int am = lane & 15;            // A row for this lane
    const int ak = (lane >> 4) * 16;     // K sub-offset within chunk
    #pragma unroll
    for (int c = 0; c < KC; ++c) {
      // A fragment: lane holds K = 32c+ak+{0..7} (halves 0-7) and
      //             K = 32c+ak+8+{0..7} (halves 8-15), row am.
      const float* ar0 = &sRawA[buf][am][32 * c + ak];
      const float* ar1 = &sRawA[buf][am][32 * c + ak + 8];
      v16h av;
      #pragma unroll
      for (int j = 0; j < 8; ++j) {
        av[j]     = (_Float16)ar0[j];
        av[j + 8] = (_Float16)ar1[j];
      }
      #pragma unroll
      for (int u = 0; u < TPW; ++u) {
        const int t = wave + u * 8;
        const v16h bv = *(const v16h*)&sB[c][t][lane][0];
        acc[u] = __builtin_amdgcn_wmma_f32_16x16x32_f16(
            false, av, false, bv, (short)0, acc[u], false, false);
      }
    }

    // Epilogue. D layout: VGPR r holds row m=r+8*(lane/16), col n=tile*16+lane%16.
    const int mBase = (lane >> 4) * 8;
    const int nLo   = lane & 15;
    #pragma unroll
    for (int u = 0; u < TPW; ++u) {
      const int t = wave + u * 8;
      const int n = t * 16 + nLo;
      #pragma unroll
      for (int r = 0; r < 8; ++r) {
        const int m = mBase + r;
        float val = acc[u][r] + bias[n];
        if constexpr (MODE == GMODE_RELU) val = val > 0.f ? val : 0.f;
        if constexpr (MODE == GMODE_LN) {
          val += resid[(size_t)(row0 + m) * NC + n];
          sO[m][n] = val;
        } else {
          out[(size_t)(row0 + m) * NC + n] = val;
        }
      }
    }

    if constexpr (MODE == GMODE_LN) {
      __syncthreads();
      const int r  = tid >> 4;   // 16 rows, 16 threads per row
      const int cs = tid & 15;
      float p1 = 0.f, p2 = 0.f;
      #pragma unroll
      for (int cc = 0; cc < NC / 16; ++cc) {
        const float xv = sO[r][cs + cc * 16];
        p1 += xv; p2 += xv * xv;
      }
      sSum[r][cs] = p1; sSq[r][cs] = p2;
      __syncthreads();
      if (cs == 0) {
        float t1 = 0.f, t2 = 0.f;
        #pragma unroll
        for (int j = 0; j < 16; ++j) { t1 += sSum[r][j]; t2 += sSq[r][j]; }
        const float mean = t1 / (float)NC;
        const float var  = t2 / (float)NC - mean * mean;
        sMean[r] = mean;
        sRstd[r] = rsqrtf(var + 1e-5f);
      }
      __syncthreads();
      #pragma unroll
      for (int cc = 0; cc < NC / 16; ++cc) {
        const int n = cs + cc * 16;
        const float xv = (sO[r][n] - sMean[r]) * sRstd[r] * lng[n] + lnb[n];
        out[(size_t)(row0 + r) * NC + n] = xv;
      }
    }
    buf ^= 1;
  }
}

// ---------------------------------------------------------------------------
// Edge-parallel attention (segment softmax via ordered-uint atomicMax + f32
// atomicAdd -> global_atomic_max_u32 / global_atomic_add_f32)
// ---------------------------------------------------------------------------
__device__ __forceinline__ unsigned ordEnc(float f) {
  const unsigned b = __float_as_uint(f);
  return b ^ (unsigned)(((int)b >> 31) | 0x80000000);
}
__device__ __forceinline__ float ordDec(unsigned u) {
  const unsigned b = (u & 0x80000000u) ? (u ^ 0x80000000u) : ~u;
  return __uint_as_float(b);
}

__global__ void gtn_init_md(unsigned* __restrict__ mOrd, float* __restrict__ den, int n) {
  const int i = blockIdx.x * blockDim.x + threadIdx.x;
  if (i < n) { mOrd[i] = ordEnc(-__builtin_inff()); den[i] = 0.f; }
}

__global__ void gtn_fill0(float* __restrict__ p, int n) {
  const int i = blockIdx.x * blockDim.x + threadIdx.x;
  if (i < n) p[i] = 0.f;
}

__global__ void gtn_edge_score(const float* __restrict__ q, const float* __restrict__ k,
                               const int* __restrict__ src, const int* __restrict__ dst,
                               float* __restrict__ sc, unsigned* __restrict__ mOrd, int EH) {
  const int i = blockIdx.x * blockDim.x + threadIdx.x;
  if (i >= EH) return;
  const int e = i >> 3, hh = i & 7;
  const int sN = src[e], dN = dst[e];
  const float* qp = q + (size_t)dN * kD + hh * kDH;
  const float* kp = k + (size_t)sN * kD + hh * kDH;
  float acc = 0.f;
  #pragma unroll
  for (int j = 0; j < kDH; ++j) acc += qp[j] * kp[j];
  acc *= 0.25f;  // 1/sqrt(DH)
  sc[i] = acc;
  atomicMax(&mOrd[dN * kH + hh], ordEnc(acc));
}

__global__ void gtn_edge_exp(float* __restrict__ sc, const unsigned* __restrict__ mOrd,
                             const int* __restrict__ dst, float* __restrict__ den, int EH) {
  const int i = blockIdx.x * blockDim.x + threadIdx.x;
  if (i >= EH) return;
  const int e = i >> 3, hh = i & 7;
  const int dN = dst[e];
  const float ex = expf(sc[i] - ordDec(mOrd[dN * kH + hh]));
  sc[i] = ex;  // overwrite score with exp in place
  atomicAdd(&den[dN * kH + hh], ex);
}

__global__ void gtn_edge_agg(const float* __restrict__ ex, const float* __restrict__ den,
                             const float* __restrict__ v, const int* __restrict__ src,
                             const int* __restrict__ dst, float* __restrict__ agg, int EH) {
  const int i = blockIdx.x * blockDim.x + threadIdx.x;
  if (i >= EH) return;
  const int e = i >> 3, hh = i & 7;
  const int sN = src[e], dN = dst[e];
  const float alpha = ex[i] / (den[dN * kH + hh] + 1e-9f);
  const float* vp = v + (size_t)sN * kD + hh * kDH;
  float* ap = agg + (size_t)dN * kD + hh * kDH;
  #pragma unroll
  for (int j = 0; j < kDH; ++j) atomicAdd(&ap[j], alpha * vp[j]);
}

// ---------------------------------------------------------------------------
// Predictor MLP: z = h[a]*h[b]; leaky_relu(z@Wp1+bp1) @ Wp2 + bp2
// ---------------------------------------------------------------------------
__global__ __launch_bounds__(64) void gtn_predict(
    const float* __restrict__ h, const int* __restrict__ ia, const int* __restrict__ ib,
    const float* __restrict__ Wp1, const float* __restrict__ bp1,
    const float* __restrict__ Wp2, const float* __restrict__ bp2,
    float* __restrict__ out)
{
  __shared__ float z[kD];
  __shared__ float red[64];
  const int p = blockIdx.x, t = threadIdx.x;
  const size_t a = (size_t)ia[p] * kD, b = (size_t)ib[p] * kD;
  z[t]      = h[a + t]      * h[b + t];
  z[t + 64] = h[a + t + 64] * h[b + t + 64];
  __syncthreads();
  float acc = bp1[t];
  #pragma unroll 8
  for (int k2 = 0; k2 < kD; ++k2) acc += z[k2] * Wp1[k2 * 64 + t];
  acc = acc > 0.f ? acc : 0.2f * acc;  // leaky relu
  red[t] = acc * Wp2[t];
  for (int s2 = 32; s2 > 0; s2 >>= 1) {
    __syncthreads();
    if (t < s2) red[t] += red[t + s2];
  }
  __syncthreads();
  if (t == 0) out[p] = red[0] + bp2[0];
}

// ---------------------------------------------------------------------------
extern "C" void kernel_launch(void* const* d_in, const int* in_sizes, int n_in,
                              void* d_out, int out_size, void* d_ws, size_t ws_size,
                              hipStream_t stream) {
  const float* x    = (const float*)d_in[0];
  const int*   src  = (const int*)d_in[1];
  const int*   dst  = (const int*)d_in[2];
  const int*   posS = (const int*)d_in[3];
  const int*   posD = (const int*)d_in[4];
  const int*   negS = (const int*)d_in[5];
  const int*   negD = (const int*)d_in[6];
  const float* W_in = (const float*)d_in[7];
  const float* b_in = (const float*)d_in[8];
  const float* Wq   = (const float*)d_in[9];
  const float* bq   = (const float*)d_in[10];
  const float* Wk   = (const float*)d_in[11];
  const float* bk   = (const float*)d_in[12];
  const float* Wv   = (const float*)d_in[13];
  const float* bv   = (const float*)d_in[14];
  const float* Wo   = (const float*)d_in[15];
  const float* bo   = (const float*)d_in[16];
  const float* ln1g = (const float*)d_in[17];
  const float* ln1b = (const float*)d_in[18];
  const float* Wf1  = (const float*)d_in[19];
  const float* bf1  = (const float*)d_in[20];
  const float* Wf2  = (const float*)d_in[21];
  const float* bf2  = (const float*)d_in[22];
  const float* ln2g = (const float*)d_in[23];
  const float* ln2b = (const float*)d_in[24];
  const float* Wp1  = (const float*)d_in[25];
  const float* bp1  = (const float*)d_in[26];
  const float* Wp2  = (const float*)d_in[27];
  const float* bp2  = (const float*)d_in[28];

  float* outv = (float*)d_out;
  float* h    = outv + 2 * kP;   // h lives directly in the output buffer

  // Workspace layout (floats): q | k | v | sc(E*H) | mOrd(N*H) | den(N*H)
  float*    ws   = (float*)d_ws;
  float*    q    = ws;
  float*    kbuf = ws + (size_t)kN * kD;
  float*    vbuf = ws + 2 * (size_t)kN * kD;
  float*    sc   = ws + 3 * (size_t)kN * kD;
  unsigned* mOrd = (unsigned*)(sc + (size_t)kE * kH);
  float*    den  = (float*)(mOrd + (size_t)kN * kH);
  float*    agg  = q;     // q is dead after edge_score
  float*    ffh  = kbuf;  // k+v contiguous region (N*FF) dead after edge_agg

  const dim3 gemmGrid((kN + 127) / 128);
  const int  EH = kE * kH;
  const dim3 ehGrid((EH + 255) / 256);
  const dim3 ndGrid((kN * kD + 255) / 256);
  const dim3 nhGrid((kN * kH + 255) / 256);

  // h = x @ W_in + b_in
  gtn_gemm<kD, kD, GMODE_BIAS><<<gemmGrid, 256, 0, stream>>>(
      x, W_in, b_in, nullptr, nullptr, nullptr, h, kN);

  for (int l = 0; l < kL; ++l) {
    const int* sl = src + (size_t)l * kE;
    const int* dl = dst + (size_t)l * kE;

    gtn_gemm<kD, kD, GMODE_BIAS><<<gemmGrid, 256, 0, stream>>>(
        h, Wq + (size_t)l * kD * kD, bq + (size_t)l * kD,
        nullptr, nullptr, nullptr, q, kN);
    gtn_gemm<kD, kD, GMODE_BIAS><<<gemmGrid, 256, 0, stream>>>(
        h, Wk + (size_t)l * kD * kD, bk + (size_t)l * kD,
        nullptr, nullptr, nullptr, kbuf, kN);
    gtn_gemm<kD, kD, GMODE_BIAS><<<gemmGrid, 256, 0, stream>>>(
        h, Wv + (size_t)l * kD * kD, bv + (size_t)l * kD,
        nullptr, nullptr, nullptr, vbuf, kN);

    gtn_init_md<<<nhGrid, 256, 0, stream>>>(mOrd, den, kN * kH);
    gtn_edge_score<<<ehGrid, 256, 0, stream>>>(q, kbuf, sl, dl, sc, mOrd, EH);
    gtn_edge_exp<<<ehGrid, 256, 0, stream>>>(sc, mOrd, dl, den, EH);
    gtn_fill0<<<ndGrid, 256, 0, stream>>>(agg, kN * kD);
    gtn_edge_agg<<<ehGrid, 256, 0, stream>>>(sc, den, vbuf, sl, dl, agg, EH);

    // h = LN(h + agg@Wo + bo)
    gtn_gemm<kD, kD, GMODE_LN><<<gemmGrid, 256, 0, stream>>>(
        agg, Wo + (size_t)l * kD * kD, bo + (size_t)l * kD,
        h, ln1g + (size_t)l * kD, ln1b + (size_t)l * kD, h, kN);
    // ffh = relu(h @ Wf1 + bf1)
    gtn_gemm<kD, kFF, GMODE_RELU><<<gemmGrid, 256, 0, stream>>>(
        h, Wf1 + (size_t)l * kD * kFF, bf1 + (size_t)l * kFF,
        nullptr, nullptr, nullptr, ffh, kN);
    // h = LN(h + ffh @ Wf2 + bf2)
    gtn_gemm<kFF, kD, GMODE_LN><<<gemmGrid, 256, 0, stream>>>(
        ffh, Wf2 + (size_t)l * kFF * kD, bf2 + (size_t)l * kD,
        h, ln2g + (size_t)l * kD, ln2b + (size_t)l * kD, h, kN);
  }

  gtn_predict<<<dim3(kP), 64, 0, stream>>>(h, posS, posD, Wp1, bp1, Wp2, bp2, outv);
  gtn_predict<<<dim3(kP), 64, 0, stream>>>(h, negS, negD, Wp1, bp1, Wp2, bp2, outv + kP);

  (void)in_sizes; (void)n_in; (void)out_size; (void)ws_size;
}
